// attentive_TCN_18760417149077
// MI455X (gfx1250) — compile-verified, matrix operand order checked by reading, not run on hardware
//
#include <hip/hip_runtime.h>
#include <hip/hip_fp16.h>
#include <stdint.h>

// ---------------------------------------------------------------------------
// Shapes (compile-time, from the reference)
// ---------------------------------------------------------------------------
#define BB 16
#define TT 2048
#define FF 128
#define UU 256
#define MM (BB * TT)   // 32768 total rows

#define VLD 264        // LDS leading dim for v block (256 + 8 pad, bank-spread)
#define KLD 40         // LDS leading dim for kT block (32 + 8 pad, bank-spread)

typedef __attribute__((ext_vector_type(16))) _Float16 v16h;
typedef __attribute__((ext_vector_type(8)))  _Float16 v8h;
typedef __attribute__((ext_vector_type(8)))  float    v8f;

// GCC-style int4 to match the async-LDS builtin's parameter type exactly
// (diagnostic showed: '__attribute__((__vector_size__(4 * sizeof(int)))) int __device__ *')
typedef int b128_t __attribute__((vector_size(16)));
typedef b128_t __attribute__((address_space(1))) gb128_t;   // global (__device__)
typedef b128_t __attribute__((address_space(3))) lb128_t;   // LDS (__shared__)

#if __has_builtin(__builtin_amdgcn_global_load_async_to_lds_b128) && \
    __has_builtin(__builtin_amdgcn_s_wait_asynccnt)
#define USE_ASYNC 1
#else
#define USE_ASYNC 0
#endif

// ---------------------------------------------------------------------------
// WMMA + fragment helpers
// ---------------------------------------------------------------------------
__device__ __forceinline__ v8f wmma_f16(v16h a, v16h b, v8f c) {
    return __builtin_amdgcn_wmma_f32_16x16x32_f16(
        /*neg_a=*/false, a, /*neg_b=*/false, b,
        /*c_mod=*/(short)0, c, /*reuse_a=*/false, /*reuse_b=*/false);
}

// Load one 16x32 f16 A/B fragment from a row-major f16 matrix with leading
// dim `ld`, tile origin at `base`. ISA layout (CDNA5 ISA): lane L holds row
// L%16; lanes 0-15 hold K chunks [0..7],[16..23]; lanes 16-31 hold
// [8..15],[24..31]. Each chunk = 8 contiguous f16 = one 16-byte load.
__device__ __forceinline__ v16h load_frag_f16(const _Float16* base, int ld, int lane) {
    const int r    = lane & 15;
    const int koff = (lane >> 4) << 3;   // 0 or 8
    const _Float16* p = base + r * ld + koff;
    v8h lo = *(const v8h*)(p);
    v8h hi = *(const v8h*)(p + 16);
    return __builtin_shufflevector(lo, hi, 0,1,2,3,4,5,6,7,8,9,10,11,12,13,14,15);
}

__device__ __forceinline__ float sigmoidf_fast(float x) {
    return 1.0f / (1.0f + __expf(-x));
}

// 16-byte global -> LDS copy: async (ASYNCcnt-tracked DMA into LDS) when the
// toolchain exposes the gfx1250 builtin, else a plain load+store pair.
__device__ __forceinline__ void copy16_g2l(_Float16* lds_dst, const _Float16* gsrc) {
#if USE_ASYNC
    __builtin_amdgcn_global_load_async_to_lds_b128(
        (gb128_t*)(uintptr_t)gsrc,
        (lb128_t*)(uintptr_t)lds_dst,
        /*imm offset=*/0, /*cpol=*/0);
#else
    *(v8h*)lds_dst = *(const v8h*)gsrc;
#endif
}

__device__ __forceinline__ void wait_async_upto16() {
#if USE_ASYNC
    __builtin_amdgcn_s_wait_asynccnt(16);   // current buffer complete, next in flight
#endif
}
__device__ __forceinline__ void wait_async_zero() {
#if USE_ASYNC
    __builtin_amdgcn_s_wait_asynccnt(0);
#endif
}

// ---------------------------------------------------------------------------
// Kernel 1: x (f32) -> xh (f16), elementwise
// ---------------------------------------------------------------------------
__global__ void cvt_x_kernel(const float* __restrict__ x,
                             _Float16* __restrict__ xh, int n) {
    int i = blockIdx.x * blockDim.x + threadIdx.x;
    if (i < n) xh[i] = (_Float16)x[i];
}

// ---------------------------------------------------------------------------
// Kernel 2: W[F][U] (f32) -> Wt[U][F] (f16), transposed for B-fragment loads.
// ---------------------------------------------------------------------------
__global__ void cvt_w_kernel(const float* __restrict__ Wq,
                             const float* __restrict__ Wv,
                             const float* __restrict__ Wk,
                             _Float16* __restrict__ wt) {
    int i = blockIdx.x * blockDim.x + threadIdx.x;
    int z = blockIdx.z;
    const float* W = (z == 0) ? Wq : (z == 1) ? Wv : Wk;
    if (i < FF * UU) {
        int f = i / UU;
        int u = i % UU;
        wt[(size_t)z * UU * FF + (size_t)u * FF + f] = (_Float16)W[i];
    }
}

// ---------------------------------------------------------------------------
// Kernel 3: projections  out = sigmoid(x @ W)  via WMMA.
//   q, v stored row-major f16 [M][U]; k stored transposed f16 [B][U][T].
// ---------------------------------------------------------------------------
__global__ __launch_bounds__(128) void proj_kernel(
    const _Float16* __restrict__ xh,
    const _Float16* __restrict__ wqt,
    const _Float16* __restrict__ wvt,
    const _Float16* __restrict__ wkt,
    _Float16* __restrict__ qh,
    _Float16* __restrict__ vh,
    _Float16* __restrict__ kT) {

    const int z    = blockIdx.z;
    const _Float16* Wt = (z == 0) ? wqt : (z == 1) ? wvt : wkt;
    const int row0 = blockIdx.x * 16;
    const int tid  = threadIdx.x;
    const int w    = tid >> 5;
    const int lane = tid & 31;
    const int n0   = w * 64;

    v8f acc[4];
#pragma unroll
    for (int n = 0; n < 4; ++n)
#pragma unroll
        for (int i = 0; i < 8; ++i) acc[n][i] = 0.0f;

    const _Float16* Abase = xh + (size_t)row0 * FF;
#pragma unroll
    for (int kk = 0; kk < 4; ++kk) {
        const int k0 = kk * 32;
        v16h a = load_frag_f16(Abase + k0, FF, lane);
#pragma unroll
        for (int n = 0; n < 4; ++n) {
            v16h b = load_frag_f16(Wt + (size_t)(n0 + n * 16) * FF + k0, FF, lane);
            acc[n] = wmma_f16(a, b, acc[n]);
        }
    }

    const int scol  = lane & 15;
    const int rbase = (lane >> 4) * 8;

    if (z == 2) {
        const int bat   = row0 / TT;
        const int tloc0 = (row0 % TT) + rbase;
#pragma unroll
        for (int n = 0; n < 4; ++n) {
            const int u = n0 + n * 16 + scol;
            v8h pk;
#pragma unroll
            for (int v = 0; v < 8; ++v) pk[v] = (_Float16)sigmoidf_fast(acc[n][v]);
            *(v8h*)(kT + (size_t)bat * UU * TT + (size_t)u * TT + tloc0) = pk;
        }
    } else {
        _Float16* out = (z == 0) ? qh : vh;
#pragma unroll
        for (int n = 0; n < 4; ++n) {
            const int u = n0 + n * 16 + scol;
#pragma unroll
            for (int v = 0; v < 8; ++v) {
                const int t = row0 + rbase + v;
                out[(size_t)t * UU + u] = (_Float16)sigmoidf_fast(acc[n][v]);
            }
        }
    }
}

// ---------------------------------------------------------------------------
// Cooperative stage of one 32-wide key block into LDS (double-buffered).
//   v block : 32 rows x 256 cols f16  -> vdst[32][VLD]
//   kT block: 256 rows x 32 cols f16  -> kdst[256][KLD]
// 1024 + 1024 b128 lane-transfers across 128 threads = 16 async instructions
// per wave per stage.
// ---------------------------------------------------------------------------
__device__ __forceinline__ void stage_block(
    _Float16* __restrict__ vdst, _Float16* __restrict__ kdst,
    const _Float16* __restrict__ vb, const _Float16* __restrict__ kTb,
    int s0, int tid) {
#pragma unroll
    for (int i = 0; i < 8; ++i) {
        const int id = tid + i * 128;   // 0..1023
        const int s  = id >> 5;         // 0..31
        const int c  = id & 31;         // 16B chunk
        copy16_g2l(vdst + s * VLD + c * 8, vb + (size_t)(s0 + s) * UU + c * 8);
    }
#pragma unroll
    for (int i = 0; i < 8; ++i) {
        const int id = tid + i * 128;   // 0..1023
        const int u  = id >> 2;         // 0..255
        const int c  = id & 3;          // 16B chunk
        copy16_g2l(kdst + u * KLD + c * 8, kTb + (size_t)u * TT + s0 + c * 8);
    }
}

// ---------------------------------------------------------------------------
// Kernel 4: fused causal attention, flash style, async double-buffered LDS.
//   128 threads = 4 waves; block covers 64 query rows of one batch; all waves
//   run a uniform number of key blocks (extra blocks fully masked -> exact
//   no-ops in the online-softmax recurrence) so staging can be cooperative.
//   Per key block per wave: 16 WMMAs for S = q.v^T, 16 WMMAs for O += P@k.
// ---------------------------------------------------------------------------
__global__ __launch_bounds__(128) void attn_kernel(
    const _Float16* __restrict__ qh,
    const _Float16* __restrict__ vh,
    const _Float16* __restrict__ kT,
    float* __restrict__ out) {

    __shared__ __align__(16) _Float16 vlds[2][32 * VLD];   // 33.8 KB
    __shared__ __align__(16) _Float16 klds[2][256 * KLD];  // 41.0 KB
    __shared__ __align__(16) _Float16 plds[4][16][40];     //  5.1 KB

    const int tid    = threadIdx.x;
    const int w      = tid >> 5;
    const int lane   = tid & 31;
    const int bat    = blockIdx.x / (TT / 64);
    const int t_base = (blockIdx.x % (TT / 64)) * 64;
    const int t0     = t_base + w * 16;

    const _Float16* qb  = qh + ((size_t)bat * TT + t0) * UU;
    const _Float16* vb  = vh + (size_t)bat * TT * UU;
    const _Float16* kTb = kT + (size_t)bat * UU * TT;

    // q fragments are reused by every key block -> keep in VGPRs
    v16h aq[8];
#pragma unroll
    for (int ku = 0; ku < 8; ++ku) aq[ku] = load_frag_f16(qb + ku * 32, UU, lane);

    v8f accO[16];
#pragma unroll
    for (int ut = 0; ut < 16; ++ut)
#pragma unroll
        for (int i = 0; i < 8; ++i) accO[ut][i] = 0.0f;

    float m[8], l[8];
#pragma unroll
    for (int v = 0; v < 8; ++v) { m[v] = -3.0e38f; l[v] = 0.0f; }

    const int   scol  = lane & 15;
    const int   rbase = (lane >> 4) * 8;
    const float sc    = 0.08838834764831845f;   // 1/sqrt(F=128)

    const int nb = t_base / 32 + 2;   // uniform across the block

    stage_block(vlds[0], klds[0], vb, kTb, 0, tid);

    for (int i = 0; i < nb; ++i) {
        const int cur = i & 1;
        const int s0  = i * 32;

        if (i + 1 < nb) {
            stage_block(vlds[cur ^ 1], klds[cur ^ 1], vb, kTb, s0 + 32, tid);
            wait_async_upto16();     // current buffer landed; next still in flight
        } else {
            wait_async_zero();
        }
        __syncthreads();             // whole double-buffer slice visible block-wide

        const _Float16* vl = vlds[cur];
        const _Float16* kl = klds[cur];

        // ---- S = q . v^T for two 16-col s-tiles (from LDS) ---------------
        v8f sa0, sa1;
#pragma unroll
        for (int ii = 0; ii < 8; ++ii) { sa0[ii] = 0.0f; sa1[ii] = 0.0f; }
#pragma unroll
        for (int ku = 0; ku < 8; ++ku) {
            v16h b0 = load_frag_f16(vl + ku * 32, VLD, lane);
            v16h b1 = load_frag_f16(vl + 16 * VLD + ku * 32, VLD, lane);
            sa0 = wmma_f16(aq[ku], b0, sa0);
            sa1 = wmma_f16(aq[ku], b1, sa1);
        }

        // ---- scale + strict causal mask + online softmax -----------------
        const bool needMask = (s0 + 32 > t0);   // wave-uniform
        float alpha[8];
#pragma unroll
        for (int v = 0; v < 8; ++v) {
            const int tg = t0 + rbase + v;
            float x0 = sa0[v] * sc;
            float x1 = sa1[v] * sc;
            if (needMask) {
                if (s0 + scol      >= tg) x0 = -1.0e30f;   // mask s >= t
                if (s0 + 16 + scol >= tg) x1 = -1.0e30f;
            }
            float mx = fmaxf(x0, x1);
            mx = fmaxf(mx, __shfl_xor(mx, 1));
            mx = fmaxf(mx, __shfl_xor(mx, 2));
            mx = fmaxf(mx, __shfl_xor(mx, 4));
            mx = fmaxf(mx, __shfl_xor(mx, 8));   // within the 16-lane row group
            const float mn = fmaxf(m[v], mx);
            const float e0 = __expf(x0 - mn);
            const float e1 = __expf(x1 - mn);
            float rs = e0 + e1;
            rs += __shfl_xor(rs, 1);
            rs += __shfl_xor(rs, 2);
            rs += __shfl_xor(rs, 4);
            rs += __shfl_xor(rs, 8);
            const float al = __expf(m[v] - mn);
            l[v]     = l[v] * al + rs;
            m[v]     = mn;
            alpha[v] = al;
            plds[w][rbase + v][scol]      = (_Float16)e0;
            plds[w][rbase + v][scol + 16] = (_Float16)e1;
        }

#pragma unroll
        for (int ut = 0; ut < 16; ++ut)
#pragma unroll
            for (int v = 0; v < 8; ++v) accO[ut][v] *= alpha[v];

        // ---- O += P @ k (kT slice from LDS) ------------------------------
        v16h ap = load_frag_f16(&plds[w][0][0], 40, lane);
#pragma unroll
        for (int ut = 0; ut < 16; ++ut) {
            v16h bk = load_frag_f16(kl + (ut * 16) * KLD, KLD, lane);
            accO[ut] = wmma_f16(ap, bk, accO[ut]);
        }

        __syncthreads();   // all reads of `cur` done before it is re-staged
    }

    // ---- epilogue: normalize; row t==0 is fully masked -> zeros ----------
    float rl[8];
#pragma unroll
    for (int v = 0; v < 8; ++v) rl[v] = (l[v] > 0.0f) ? (1.0f / l[v]) : 0.0f;
#pragma unroll
    for (int ut = 0; ut < 16; ++ut) {
        const int u = ut * 16 + scol;
#pragma unroll
        for (int v = 0; v < 8; ++v) {
            const int tg = t0 + rbase + v;
            const float val = (tg == 0) ? 0.0f : accO[ut][v] * rl[v];
            out[((size_t)bat * TT + tg) * UU + u] = val;
        }
    }
}

// ---------------------------------------------------------------------------
// Host-side launcher
// ---------------------------------------------------------------------------
extern "C" void kernel_launch(void* const* d_in, const int* in_sizes, int n_in,
                              void* d_out, int out_size, void* d_ws, size_t ws_size,
                              hipStream_t stream) {
    (void)in_sizes; (void)n_in; (void)out_size; (void)ws_size;

    const float* x  = (const float*)d_in[0];
    const float* Wq = (const float*)d_in[1];
    const float* Wv = (const float*)d_in[2];
    const float* Wk = (const float*)d_in[3];
    float* out = (float*)d_out;

    _Float16* ws = (_Float16*)d_ws;
    _Float16* xh = ws;                                  // [M][F]
    _Float16* wt = xh + (size_t)MM * FF;                // 3 x [U][F] (transposed)
    _Float16* qh = wt + 3 * (size_t)UU * FF;            // [M][U]
    _Float16* vh = qh + (size_t)MM * UU;                // [M][U]
    _Float16* kT = vh + (size_t)MM * UU;                // [B][U][T]

    cvt_x_kernel<<<(MM * FF + 255) / 256, 256, 0, stream>>>(x, xh, MM * FF);
    cvt_w_kernel<<<dim3((FF * UU + 255) / 256, 1, 3), 256, 0, stream>>>(Wq, Wv, Wk, wt);
    proj_kernel<<<dim3(MM / 16, 1, 3), 128, 0, stream>>>(
        xh, wt, wt + (size_t)UU * FF, wt + 2 * (size_t)UU * FF, qh, vh, kT);
    attn_kernel<<<BB * (TT / 64), 128, 0, stream>>>(qh, vh, kT, out);
}
